// CausalSelfAttention_78546361909748
// MI455X (gfx1250) — matrix operand order repaired and checked by reading, not compiled, not run on hardware
//
#include <hip/hip_runtime.h>
#include <hip/hip_bf16.h>

// ---------------------------------------------------------------------------
// Vector types matching the gfx1250 WMMA builtin signatures.
// ---------------------------------------------------------------------------
typedef _Float16 h4  __attribute__((ext_vector_type(4)));
typedef _Float16 h8  __attribute__((ext_vector_type(8)));
typedef _Float16 h16 __attribute__((ext_vector_type(16)));
typedef float    f8  __attribute__((ext_vector_type(8)));

union H16 { h16 v; h8 h[2]; };

__device__ __forceinline__ f8 wmma_f16(h16 a, h16 b, f8 c) {
  // D = A(16x32 f16) * B(32x16 f16) + C(16x16 f32)
  return __builtin_amdgcn_wmma_f32_16x16x32_f16(
      /*neg_a=*/false, a, /*neg_b=*/false, b,
      /*c_mod=*/(short)0, c, /*reuse_a=*/false, /*reuse_b=*/false);
}

// ---------------------------------------------------------------------------
// Tensor Data Mover (TDM) support: async global->LDS tile DMA, TENSORcnt.
// ---------------------------------------------------------------------------
#if __has_builtin(__builtin_amdgcn_tensor_load_to_lds) && \
    __has_builtin(__builtin_amdgcn_s_wait_tensorcnt)
#define HAVE_TDM 1
#else
#define HAVE_TDM 0
#endif

#if HAVE_TDM
typedef unsigned int u32x4 __attribute__((ext_vector_type(4)));
typedef int          i32x4 __attribute__((ext_vector_type(4)));
typedef int          i32x8 __attribute__((ext_vector_type(8)));

// Issue one TDM descriptor: copy a tileRows x tileCols tile of f16 (row-major,
// row stride = rowStride elements) from global memory into LDS (contiguous).
__device__ __forceinline__ void tdm_load_tile_f16(
    const _Float16* gsrc, void* ldsDst,
    unsigned tileCols, unsigned tileRows,
    unsigned tensorCols, unsigned tensorRows, unsigned rowStride)
{
  const unsigned long long ga = (unsigned long long)(uintptr_t)gsrc;
  // D# group 0: [1:0]=count=1 (valid), [63:32]=lds_addr,
  //             [120:64]=global_addr, [127:126]=type=2 ("image")
  u32x4 g0;
  g0.x = 1u;
  g0.y = (unsigned)(uintptr_t)ldsDst;
  g0.z = (unsigned)ga;
  g0.w = (unsigned)((ga >> 32) & 0x01FFFFFFull) | (2u << 30);
  // D# group 1: wg_mask=0, data_size=1 (2 bytes), no flags/pad,
  //             tensor_dim0/1, tile_dim0/1/2, tensor_dim0_stride.
  i32x8 g1;
  g1[0] = (int)(1u << 16);                                         // data_size=2B
  g1[1] = (int)((tensorCols & 0xFFFFu) << 16);                     // tdim0[15:0]
  g1[2] = (int)((tensorCols >> 16) | ((tensorRows & 0xFFFFu) << 16));
  g1[3] = (int)((tensorRows >> 16) | (tileCols << 16));            // tile_dim0
  g1[4] = (int)tileRows;                                           // tile_dim1 (tile_dim2=0)
  g1[5] = (int)rowStride;                                          // dim0 stride lo
  g1[6] = 0;                                                       // stride hi / dim1 stride
  g1[7] = 0;
  i32x4 gz = {0, 0, 0, 0};                                         // groups 2/3: unused (2D)
#if __clang_major__ >= 23
  i32x8 gz8 = {0, 0, 0, 0, 0, 0, 0, 0};
  __builtin_amdgcn_tensor_load_to_lds(g0, g1, gz, gz, gz8, 0);
#else
  __builtin_amdgcn_tensor_load_to_lds(g0, g1, gz, gz, 0);
#endif
}
#endif  // HAVE_TDM

// ---------------------------------------------------------------------------
// f32 -> f16 conversion (vectorized x4)
// ---------------------------------------------------------------------------
__global__ __launch_bounds__(256) void cvt_f32_f16(
    const float* __restrict__ in, _Float16* __restrict__ out, int n4) {
  int i = blockIdx.x * 256 + threadIdx.x;
  if (i < n4) {
    const float4 v = ((const float4*)in)[i];
    h4 o;
    o.x = (_Float16)v.x; o.y = (_Float16)v.y;
    o.z = (_Float16)v.z; o.w = (_Float16)v.w;
    ((h4*)out)[i] = o;
  }
}

// ---------------------------------------------------------------------------
// GEMM: C[M,N] = A[M,K] @ W[N,K]^T   (nn.Linear convention: x @ W.T)
// f16 inputs, f32 WMMA accumulation, OutT output.
// Block tile 128x128, 8 waves (2x4), wave tile 64x32 = 4x2 WMMA tiles.
// K staged through double-buffered LDS in 32-wide steps via TDM async DMA
// (wave 0 issues descriptors, TENSORcnt-synchronized), VALU fallback if the
// toolchain lacks the TDM builtin.
// ---------------------------------------------------------------------------
template <typename OutT>
__global__ __launch_bounds__(256) void gemm_xwt(
    const _Float16* __restrict__ A, const _Float16* __restrict__ W,
    OutT* __restrict__ C, int M, int N, int K)
{
#if HAVE_TDM
  __shared__ __align__(128) _Float16 As[2][128][32];
  __shared__ __align__(128) _Float16 Ws[2][128][32];
#else
  __shared__ __align__(128) _Float16 As[1][128][32];
  __shared__ __align__(128) _Float16 Ws[1][128][32];
#endif

  const int tid  = threadIdx.x;
  const int wave = tid >> 5;
  const int lane = tid & 31;
  const int l15  = lane & 15;
  const int hlf  = lane >> 4;

  const int bm = blockIdx.x * 128;
  const int bn = blockIdx.y * 128;
  const int wm = (wave >> 2) * 64;   // wave row offset within block tile
  const int wn = (wave & 3) * 32;    // wave col offset within block tile

  f8 acc[4][2];
  #pragma unroll
  for (int i = 0; i < 4; ++i)
    #pragma unroll
    for (int j = 0; j < 2; ++j)
      acc[i][j] = f8{};

  const int steps = K / 32;
  const _Float16* gAbase = A + (size_t)bm * K;
  const _Float16* gWbase = W + (size_t)bn * K;

#if HAVE_TDM
  // Prime buffer 0 with the first K-slab (one wave drives the DMA engine).
  if (wave == 0) {
    tdm_load_tile_f16(gAbase, &As[0][0][0], 32, 128, (unsigned)K, (unsigned)M, (unsigned)K);
    tdm_load_tile_f16(gWbase, &Ws[0][0][0], 32, 128, (unsigned)K, (unsigned)N, (unsigned)K);
  }
#else
  const int ldRow = tid >> 1;
  const int ldCol = (tid & 1) * 16;
  const _Float16* gA = gAbase + (size_t)ldRow * K + ldCol;
  const _Float16* gW = gWbase + (size_t)ldRow * K + ldCol;
#endif

  for (int kk = 0; kk < steps; ++kk) {
#if HAVE_TDM
    const int cur = kk & 1;
    __syncthreads();  // everyone done reading buffer cur^1 (prev iteration)
    if (wave == 0) {
      if (kk + 1 < steps) {
        const int k1 = (kk + 1) * 32;
        tdm_load_tile_f16(gAbase + k1, &As[cur ^ 1][0][0], 32, 128,
                          (unsigned)K, (unsigned)M, (unsigned)K);
        tdm_load_tile_f16(gWbase + k1, &Ws[cur ^ 1][0][0], 32, 128,
                          (unsigned)K, (unsigned)N, (unsigned)K);
        __builtin_amdgcn_s_wait_tensorcnt(2);  // current pair done (in-order)
      } else {
        __builtin_amdgcn_s_wait_tensorcnt(0);
      }
    }
    __syncthreads();  // release all waves onto buffer cur
#else
    const int cur = 0;
    const int k0 = kk * 32;
    __syncthreads();
    *(h8*)&As[0][ldRow][ldCol]     = *(const h8*)(gA + k0);
    *(h8*)&As[0][ldRow][ldCol + 8] = *(const h8*)(gA + k0 + 8);
    *(h8*)&Ws[0][ldRow][ldCol]     = *(const h8*)(gW + k0);
    *(h8*)&Ws[0][ldRow][ldCol + 8] = *(const h8*)(gW + k0 + 8);
    __syncthreads();
#endif

    // A fragment (16x32): lane l<16 holds M=l, K={0..7,16..23}; lane l+16: K={8..15,24..31}
    H16 af[4];
    #pragma unroll
    for (int tm = 0; tm < 4; ++tm) {
      const int r = wm + tm * 16 + l15;
      af[tm].h[0] = *(const h8*)&As[cur][r][hlf * 8];
      af[tm].h[1] = *(const h8*)&As[cur][r][16 + hlf * 8];
    }
    // B fragment (32x16 = W^T tile): lane l holds column N=l15 (row of W), K={hlf*8..+7, 16+hlf*8..+7}
    H16 bf[2];
    #pragma unroll
    for (int tn = 0; tn < 2; ++tn) {
      const int r = wn + tn * 16 + l15;
      bf[tn].h[0] = *(const h8*)&Ws[cur][r][hlf * 8];
      bf[tn].h[1] = *(const h8*)&Ws[cur][r][16 + hlf * 8];
    }
    #pragma unroll
    for (int tm = 0; tm < 4; ++tm)
      #pragma unroll
      for (int tn = 0; tn < 2; ++tn)
        acc[tm][tn] = wmma_f16(af[tm].v, bf[tn].v, acc[tm][tn]);
  }

  // C/D layout: lanes 0-15 -> N=l15, M=r; lanes 16-31 -> N=l15, M=8+r
  #pragma unroll
  for (int tm = 0; tm < 4; ++tm) {
    #pragma unroll
    for (int tn = 0; tn < 2; ++tn) {
      const int row0 = bm + wm + tm * 16 + hlf * 8;
      const int col  = bn + wn + tn * 16 + l15;
      #pragma unroll
      for (int r = 0; r < 8; ++r)
        C[(size_t)(row0 + r) * N + col] = (OutT)acc[tm][tn][r];
    }
  }
}

// ---------------------------------------------------------------------------
// Windowed causal attention, WINDOW == block == 16.
// One wave per (b, h, query-block). Query block qb attends only key blocks
// qb (lower-triangular incl. diag) and qb-1 (strictly upper-triangular).
// ---------------------------------------------------------------------------
__global__ __launch_bounds__(32) void attn_win16(
    const _Float16* __restrict__ Q, const _Float16* __restrict__ Km,
    const _Float16* __restrict__ V, _Float16* __restrict__ Y,
    int T, int Hn)
{
  __shared__ __align__(16) _Float16 P[16][32];  // probs: [query][key: prev 0..15 | cur 16..31]

  const int qb   = blockIdx.x;
  const int h    = blockIdx.y;
  const int b    = blockIdx.z;
  const int lane = threadIdx.x;
  const int l15  = lane & 15;
  const int hlf  = lane >> 4;

  const int    Dm      = Hn * 64;
  const size_t baseRow = (size_t)b * T;
  const int    colBase = h * 64;

  // ---- Q as A fragments (16x32 per K-half of HD=64) ----
  const _Float16* qrow = Q + (baseRow + qb * 16 + l15) * Dm + colBase;
  H16 qa0, qa1;
  qa0.h[0] = *(const h8*)(qrow + hlf * 16);
  qa0.h[1] = *(const h8*)(qrow + hlf * 16 + 8);
  qa1.h[0] = *(const h8*)(qrow + 32 + hlf * 16);
  qa1.h[1] = *(const h8*)(qrow + 32 + hlf * 16 + 8);

  const int pb = (qb > 0) ? (qb - 1) : 0;

  // ---- S = Q K^T via WMMA; B fragment column n = key row n of K ----
  f8 s_cur = f8{}, s_prev = f8{};
  {
    const _Float16* krow = Km + (baseRow + qb * 16 + l15) * Dm + colBase;
    H16 k0f, k1f;
    k0f.h[0] = *(const h8*)(krow + hlf * 16);
    k0f.h[1] = *(const h8*)(krow + hlf * 16 + 8);
    k1f.h[0] = *(const h8*)(krow + 32 + hlf * 16);
    k1f.h[1] = *(const h8*)(krow + 32 + hlf * 16 + 8);
    s_cur = wmma_f16(qa0.v, k0f.v, s_cur);
    s_cur = wmma_f16(qa1.v, k1f.v, s_cur);
  }
  {
    const _Float16* krow = Km + (baseRow + pb * 16 + l15) * Dm + colBase;
    H16 k0f, k1f;
    k0f.h[0] = *(const h8*)(krow + hlf * 16);
    k0f.h[1] = *(const h8*)(krow + hlf * 16 + 8);
    k1f.h[0] = *(const h8*)(krow + 32 + hlf * 16);
    k1f.h[1] = *(const h8*)(krow + 32 + hlf * 16 + 8);
    s_prev = wmma_f16(qa0.v, k0f.v, s_prev);
    s_prev = wmma_f16(qa1.v, k1f.v, s_prev);
  }

  // ---- mask + softmax over 32 candidate keys (row spread across 16 lanes) ----
  const float scale = 0.125f;  // 1/sqrt(64)
  #pragma unroll
  for (int r = 0; r < 8; ++r) {
    const int m = hlf * 8 + r;  // query index within block
    const int n = l15;          // key index within block
    float sc = (n <= m)           ? (float)s_cur[r]  * scale : -1e30f;
    float sp = (qb > 0 && n > m)  ? (float)s_prev[r] * scale : -1e30f;
    float mx = fmaxf(sc, sp);
    #pragma unroll
    for (int off = 1; off < 16; off <<= 1)
      mx = fmaxf(mx, __shfl_xor(mx, off, 32));
    float ec = __expf(sc - mx);
    float ep = __expf(sp - mx);
    float sum = ec + ep;
    #pragma unroll
    for (int off = 1; off < 16; off <<= 1)
      sum += __shfl_xor(sum, off, 32);
    const float inv = 1.0f / sum;
    P[m][n]      = (_Float16)(ep * inv);  // prev-block keys -> K 0..15
    P[m][16 + n] = (_Float16)(ec * inv);  // cur-block keys  -> K 16..31
  }
  __syncthreads();  // single-wave WG: acts as DS-counter fence

  // ---- re-stripe P (C layout) into an A fragment (16x32) ----
  H16 pf;
  pf.h[0] = *(const h8*)&P[l15][hlf * 16];
  pf.h[1] = *(const h8*)&P[l15][hlf * 16 + 8];

  // ---- O = P @ [V_prev; V_cur] : B fragment K rows hlf*16..+15 ----
  const int kbase = (hlf ? qb : pb) * 16;
  const _Float16* vbase = V + (baseRow + kbase) * Dm + colBase;

  f8 oacc[4];
  #pragma unroll
  for (int tn = 0; tn < 4; ++tn) oacc[tn] = f8{};

  #pragma unroll
  for (int tn = 0; tn < 4; ++tn) {
    H16 vf;
    const _Float16* vcol = vbase + tn * 16 + l15;  // column N=l15 of the hd chunk
    #pragma unroll
    for (int j = 0; j < 16; ++j)
      vf.v[j] = vcol[(size_t)j * Dm];              // stride over key rows
    oacc[tn] = wmma_f16(pf.v, vf.v, oacc[tn]);
  }

  #pragma unroll
  for (int tn = 0; tn < 4; ++tn) {
    const size_t row0 = baseRow + qb * 16 + hlf * 8;
    const int    col  = colBase + tn * 16 + l15;
    #pragma unroll
    for (int r = 0; r < 8; ++r)
      Y[(row0 + r) * Dm + col] = (_Float16)oacc[tn][r];
  }
}

// ---------------------------------------------------------------------------
// Launcher
// ---------------------------------------------------------------------------
extern "C" void kernel_launch(void* const* d_in, const int* in_sizes, int n_in,
                              void* d_out, int out_size, void* d_ws, size_t ws_size,
                              hipStream_t stream)
{
  (void)in_sizes; (void)n_in; (void)out_size; (void)ws_size;

  const int Bv = 2, T = 2048, Dm = 1024, Hn = 16;
  const int M = Bv * T;  // 4096

  const float* x  = (const float*)d_in[0];
  const float* Wq = (const float*)d_in[1];
  const float* Wk = (const float*)d_in[2];
  const float* Wv = (const float*)d_in[3];
  const float* Wo = (const float*)d_in[4];
  float* out = (float*)d_out;

  _Float16* p   = (_Float16*)d_ws;
  _Float16* xh  = p; p += (size_t)M * Dm;
  _Float16* wqh = p; p += (size_t)Dm * Dm;
  _Float16* wkh = p; p += (size_t)Dm * Dm;
  _Float16* wvh = p; p += (size_t)Dm * Dm;
  _Float16* woh = p; p += (size_t)Dm * Dm;
  _Float16* qh  = p; p += (size_t)M * Dm;
  _Float16* kh  = p; p += (size_t)M * Dm;
  _Float16* vh  = p; p += (size_t)M * Dm;
  _Float16* yh  = p; p += (size_t)M * Dm;

  auto cvt = [&](const float* src, _Float16* dst, int n) {
    const int n4 = n / 4;
    cvt_f32_f16<<<dim3((n4 + 255) / 256), dim3(256), 0, stream>>>(src, dst, n4);
  };
  cvt(x,  xh,  M * Dm);
  cvt(Wq, wqh, Dm * Dm);
  cvt(Wk, wkh, Dm * Dm);
  cvt(Wv, wvh, Dm * Dm);
  cvt(Wo, woh, Dm * Dm);

  dim3 gg(M / 128, Dm / 128);
  gemm_xwt<_Float16><<<gg, dim3(256), 0, stream>>>(xh, wqh, qh, M, Dm, Dm);
  gemm_xwt<_Float16><<<gg, dim3(256), 0, stream>>>(xh, wkh, kh, M, Dm, Dm);
  gemm_xwt<_Float16><<<gg, dim3(256), 0, stream>>>(xh, wvh, vh, M, Dm, Dm);

  attn_win16<<<dim3(T / 16, Hn, Bv), dim3(32), 0, stream>>>(qh, kh, vh, yh, T, Hn);

  gemm_xwt<float><<<gg, dim3(256), 0, stream>>>(yh, woh, out, M, Dm, Dm);
}
